// ContrastiveAttentionCompensation_46789373723081
// MI455X (gfx1250) — matrix operand order repaired and checked by reading, not compiled, not run on hardware
//
#include <hip/hip_runtime.h>
#include <cmath>

// ---------------------------------------------------------------------------
// ContrastiveAttentionCompensation for MI455X (gfx1250, wave32, WMMA)
// q = h1@Wq^T+bq ; k = h2@Wk^T+bk ; attn = q@k^T/sqrt(D)
// fused1 = softmax_row(attn)@k + q ; fused2 = softmax_col(attn)^T@q + k
// Outputs concatenated: fused1 [N,D] | fused2 [M,D] | attn [N,M]  (fp32)
// ---------------------------------------------------------------------------

typedef __attribute__((ext_vector_type(16))) __bf16 bf16x16;
typedef __attribute__((ext_vector_type(8)))  __bf16 bf16x8;
typedef __attribute__((ext_vector_type(8)))  float  f32x8;

static constexpr int NROW = 8192;   // N
static constexpr int MROW = 8192;   // M
static constexpr int DDIM = 512;    // D

// ---- WMMA wrapper: D = A(16x32 bf16) * B(32x16 bf16) + C(16x16 f32) -------
__device__ __forceinline__ f32x8 wmma_bf16(bf16x16 a, bf16x16 b, f32x8 c) {
  return __builtin_amdgcn_wmma_f32_16x16x32_bf16(
      /*neg_a=*/false, a, /*neg_b=*/false, b,
      /*c_mod=*/(short)0, c, /*reuse_a=*/false, /*reuse_b=*/false);
}

// ---- A fragment (16x32) from row-major f32, converted to bf16 -------------
__device__ __forceinline__ bf16x16 a_frag_f32(const float* __restrict__ base,
                                              int ld, int row0, int k0, int lane) {
  int r  = row0 + (lane & 15);
  int kb = k0 + ((lane >> 4) << 3);
  const float* p = base + (size_t)r * ld + kb;
  f32x8 lo = *(const f32x8*)p;          // K = kb..kb+7
  f32x8 hi = *(const f32x8*)(p + 16);   // K = kb+16..kb+23
  bf16x16 out;
#pragma unroll
  for (int i = 0; i < 8; ++i) { out[i] = (__bf16)lo[i]; out[i + 8] = (__bf16)hi[i]; }
  return out;
}

// ---- A fragment (16x32) from row-major bf16 -------------------------------
__device__ __forceinline__ bf16x16 a_frag_bf16(const __bf16* __restrict__ base,
                                               int ld, int row0, int k0, int lane) {
  int r  = row0 + (lane & 15);
  int kb = k0 + ((lane >> 4) << 3);
  const __bf16* p = base + (size_t)r * ld + kb;
  bf16x8 lo = *(const bf16x8*)p;
  bf16x8 hi = *(const bf16x8*)(p + 16);
  bf16x16 out;
#pragma unroll
  for (int i = 0; i < 8; ++i) { out[i] = lo[i]; out[i + 8] = hi[i]; }
  return out;
}

// ---- B fragment (32x16): column n holds row n of a K-contiguous matrix ----
// Bmat[k][n] = Wrow[n][k]  (e.g. W for x@W^T, or k_bf16 for q@k^T, or kT)
__device__ __forceinline__ bf16x16 b_frag_f32(const float* __restrict__ base,
                                              int ld, int k0, int col0, int lane) {
  int c  = col0 + (lane & 15);
  int kb = k0 + ((lane >> 4) << 3);
  const float* p = base + (size_t)c * ld + kb;
  f32x8 lo = *(const f32x8*)p;
  f32x8 hi = *(const f32x8*)(p + 16);
  bf16x16 out;
#pragma unroll
  for (int i = 0; i < 8; ++i) { out[i] = (__bf16)lo[i]; out[i + 8] = (__bf16)hi[i]; }
  return out;
}

__device__ __forceinline__ bf16x16 b_frag_bf16(const __bf16* __restrict__ base,
                                               int ld, int k0, int col0, int lane) {
  int c  = col0 + (lane & 15);
  int kb = k0 + ((lane >> 4) << 3);
  const __bf16* p = base + (size_t)c * ld + kb;
  bf16x8 lo = *(const bf16x8*)p;
  bf16x8 hi = *(const bf16x8*)(p + 16);
  bf16x16 out;
#pragma unroll
  for (int i = 0; i < 8; ++i) { out[i] = lo[i]; out[i + 8] = hi[i]; }
  return out;
}

// ---------------------------------------------------------------------------
// Projection: O[n,d] = H[n,:] . W[d,:] + b[d]   (Linear: H @ W^T + b)
// Also writes bf16 copy and bf16 transpose for downstream WMMA B-operands.
// Block: 256 thr = 8 waves, tile 64 rows x 512 cols (full D). Grid: rows/64.
// ---------------------------------------------------------------------------
__global__ __launch_bounds__(256) void proj_kernel(
    const float* __restrict__ H, const float* __restrict__ W,
    const float* __restrict__ bias, float* __restrict__ Of32,
    __bf16* __restrict__ Obf, __bf16* __restrict__ ObfT, int nRows) {
  const int D = DDIM;
  const int row0 = blockIdx.x * 64;
  const int wave = threadIdx.x >> 5;
  const int lane = threadIdx.x & 31;
  const int col0 = wave * 64;

  f32x8 acc[4][4] = {};
  for (int k0 = 0; k0 < D; k0 += 32) {
    bf16x16 a[4], b[4];
#pragma unroll
    for (int mi = 0; mi < 4; ++mi) a[mi] = a_frag_f32(H, D, row0 + mi * 16, k0, lane);
#pragma unroll
    for (int ni = 0; ni < 4; ++ni) b[ni] = b_frag_f32(W, D, k0, col0 + ni * 16, lane);
#pragma unroll
    for (int mi = 0; mi < 4; ++mi)
#pragma unroll
      for (int ni = 0; ni < 4; ++ni) acc[mi][ni] = wmma_bf16(a[mi], b[ni], acc[mi][ni]);
  }

  const int rbase = (lane >> 4) * 8;
#pragma unroll
  for (int mi = 0; mi < 4; ++mi)
#pragma unroll
    for (int ni = 0; ni < 4; ++ni) {
      int col = col0 + ni * 16 + (lane & 15);
      float bv = bias[col];
#pragma unroll
      for (int v = 0; v < 8; ++v) {
        int row = row0 + mi * 16 + rbase + v;
        float val = acc[mi][ni][v] + bv;
        Of32[(size_t)row * D + col] = val;
        __bf16 h = (__bf16)val;
        Obf[(size_t)row * D + col]       = h;
        ObfT[(size_t)col * nRows + row]  = h;   // transpose for fused B-operand
      }
    }
}

// ---------------------------------------------------------------------------
// attn[n,m] = (q[n,:].k[m,:]) / sqrt(D).  A=q_bf16 rows, B=k_bf16 rows (K-contig)
// Grid: (N/64, M/512). Block tile 64 x 512.
// ---------------------------------------------------------------------------
__global__ __launch_bounds__(256) void attn_kernel(
    const __bf16* __restrict__ qb, const __bf16* __restrict__ kb,
    float* __restrict__ attn) {
  const int D = DDIM;
  const int row0 = blockIdx.x * 64;
  const int wave = threadIdx.x >> 5;
  const int lane = threadIdx.x & 31;
  const int col0 = blockIdx.y * 512 + wave * 64;

  f32x8 acc[4][4] = {};
  for (int k0 = 0; k0 < D; k0 += 32) {
    bf16x16 a[4], b[4];
#pragma unroll
    for (int mi = 0; mi < 4; ++mi) a[mi] = a_frag_bf16(qb, D, row0 + mi * 16, k0, lane);
#pragma unroll
    for (int ni = 0; ni < 4; ++ni) b[ni] = b_frag_bf16(kb, D, k0, col0 + ni * 16, lane);
#pragma unroll
    for (int mi = 0; mi < 4; ++mi)
#pragma unroll
      for (int ni = 0; ni < 4; ++ni) acc[mi][ni] = wmma_bf16(a[mi], b[ni], acc[mi][ni]);
  }

  const float scale = 0.044194173824159216f; // 1/sqrt(512)
  const int rbase = (lane >> 4) * 8;
#pragma unroll
  for (int mi = 0; mi < 4; ++mi)
#pragma unroll
    for (int ni = 0; ni < 4; ++ni) {
      int col = col0 + ni * 16 + (lane & 15);
#pragma unroll
      for (int v = 0; v < 8; ++v) {
        int row = row0 + mi * 16 + rbase + v;
        attn[(size_t)row * MROW + col] = acc[mi][ni][v] * scale;
      }
    }
}

// ---------------------------------------------------------------------------
// Online softmax stats: per-row max & 1/sum(exp).  One block per row.
// ---------------------------------------------------------------------------
__global__ __launch_bounds__(256) void rowstats_kernel(
    const float* __restrict__ attn, float* __restrict__ rmax, float* __restrict__ rrinv) {
  const int row = blockIdx.x, tid = threadIdx.x;
  float m = -INFINITY, l = 0.f;
  for (int j = tid; j < MROW; j += 256) {
    if (j + 4096 < MROW)
      __builtin_prefetch((const void*)(attn + (size_t)row * MROW + j + 4096), 0, 0);
    float x = attn[(size_t)row * MROW + j];
    float mn = fmaxf(m, x);
    l = l * __expf(m - mn) + __expf(x - mn);
    m = mn;
  }
  __shared__ float sm[256], sl[256];
  sm[tid] = m; sl[tid] = l;
  __syncthreads();
  for (int s = 128; s > 0; s >>= 1) {
    if (tid < s) {
      float m2 = sm[tid + s], l2 = sl[tid + s];
      float mn = fmaxf(sm[tid], m2);
      sl[tid] = sl[tid] * __expf(sm[tid] - mn) + l2 * __expf(m2 - mn);
      sm[tid] = mn;
    }
    __syncthreads();
  }
  if (tid == 0) { rmax[row] = sm[0]; rrinv[row] = 1.f / sl[0]; }
}

// Per-column stats: thread owns one column; coalesced row sweeps.
__global__ __launch_bounds__(256) void colstats_kernel(
    const float* __restrict__ attn, float* __restrict__ cmax, float* __restrict__ crinv) {
  const int col = blockIdx.x * 256 + threadIdx.x;
  float m = -INFINITY, l = 0.f;
  for (int n = 0; n < NROW; ++n) {
    if (n + 16 < NROW)
      __builtin_prefetch((const void*)(attn + (size_t)(n + 16) * MROW + col), 0, 0);
    float x = attn[(size_t)n * MROW + col];
    float mn = fmaxf(m, x);
    l = l * __expf(m - mn) + __expf(x - mn);
    m = mn;
  }
  cmax[col] = m; crinv[col] = 1.f / l;
}

// ---------------------------------------------------------------------------
// Fused GEMMs. out[r, d] = sum_k P[r, k] * BT[d, k] + resid[r, d]
//  transposed==0 (fused1): P[r,k] = exp(attn[r,k]-rmax[r])*rrinv[r], BT=kT
//  transposed==1 (fused2): P[r,k] = exp(attn[k,r]-cmax[r])*crinv[r], BT=qT
// attn tiles staged HBM->LDS with double-buffered ASYNC copies (ASYNCcnt),
// then exp-normalized into a bf16 LDS A-tile shared by the 8 waves.
// ---------------------------------------------------------------------------
__global__ __launch_bounds__(256) void fused_kernel(
    const float* __restrict__ attn, const float* __restrict__ smax,
    const float* __restrict__ srinv, const __bf16* __restrict__ BT,
    const float* __restrict__ resid, float* __restrict__ out, int transposed) {
  const int D = DDIM, K = MROW, STR = 40;           // STR: LDS row stride (bf16)
  __shared__ __align__(16) float Af[2][64 * 32];    // raw f32 tiles, 8KB each
  __shared__ __align__(16) __bf16 As[64 * STR];     // softmax-prob A tile (bf16)

  const int row0 = blockIdx.x * 64;
  const int tid  = threadIdx.x;
  const int wave = tid >> 5, lane = tid & 31;
  const int col0 = wave * 64;

  // Issue the async HBM->LDS copy of one 64x32 (or 32x64) f32 tile.
  // 512 B128 chunks, 2 per thread; both sides 16B aligned.
  auto issue_tile = [&](int k0, int buf) {
#pragma unroll
    for (int i = 0; i < 2; ++i) {
      int cc = i * 256 + tid;
      unsigned long long gaddr;
      unsigned ldsoff;
      if (!transposed) {
        int r = cc >> 3, c4 = cc & 7;   // 64 rows x 8 chunks of 4 floats
        gaddr  = (unsigned long long)(attn + (size_t)(row0 + r) * MROW + (k0 + c4 * 4));
        ldsoff = (unsigned)(size_t)&Af[buf][r * 32 + c4 * 4];
      } else {
        int r = cc >> 4, c4 = cc & 15;  // 32 attn-rows x 16 chunks of 4 floats
        gaddr  = (unsigned long long)(attn + (size_t)(k0 + r) * MROW + (row0 + c4 * 4));
        ldsoff = (unsigned)(size_t)&Af[buf][r * 64 + c4 * 4];
      }
      asm volatile("global_load_async_to_lds_b128 %0, %1, off"
                   :: "v"(ldsoff), "v"(gaddr) : "memory");
    }
  };

  issue_tile(0, 0);                       // prologue: tile 0 in flight

  f32x8 acc[4][4] = {};
  for (int k0 = 0; k0 < K; k0 += 32) {
    const int cur = (k0 >> 5) & 1, nxt = cur ^ 1;
    asm volatile("s_wait_asynccnt 0" ::: "memory");   // this wave's copies done
    __syncthreads();                                  // whole tile resident
    if (k0 + 32 < K) issue_tile(k0 + 32, nxt);        // overlap next copy

    // exp-normalize current f32 tile into bf16 A tile (transpose if needed)
    if (!transposed) {
#pragma unroll
      for (int i = 0; i < 8; ++i) {
        int e = i * 256 + tid, r = e >> 5, c = e & 31;
        int gr = row0 + r;
        As[r * STR + c] = (__bf16)(__expf(Af[cur][r * 32 + c] - smax[gr]) * srinv[gr]);
      }
    } else {
#pragma unroll
      for (int i = 0; i < 8; ++i) {
        int e = i * 256 + tid, r = e & 63, c = e >> 6;  // r: out-row(m), c: attn-row(n)
        int gm = row0 + r;
        As[r * STR + c] = (__bf16)(__expf(Af[cur][c * 64 + r] - smax[gm]) * srinv[gm]);
      }
    }
    __syncthreads();

    bf16x16 a[4], b[4];
#pragma unroll
    for (int mi = 0; mi < 4; ++mi) {
      int r  = mi * 16 + (lane & 15);
      int kb = (lane >> 4) << 3;
      const __bf16* p = &As[r * STR + kb];
      bf16x8 lo = *(const bf16x8*)p;
      bf16x8 hi = *(const bf16x8*)(p + 16);
#pragma unroll
      for (int t = 0; t < 8; ++t) { a[mi][t] = lo[t]; a[mi][t + 8] = hi[t]; }
    }
#pragma unroll
    for (int ni = 0; ni < 4; ++ni) b[ni] = b_frag_bf16(BT, K, k0, col0 + ni * 16, lane);
#pragma unroll
    for (int mi = 0; mi < 4; ++mi)
#pragma unroll
      for (int ni = 0; ni < 4; ++ni) acc[mi][ni] = wmma_bf16(a[mi], b[ni], acc[mi][ni]);
  }

  const int rbase = (lane >> 4) * 8;
#pragma unroll
  for (int mi = 0; mi < 4; ++mi)
#pragma unroll
    for (int ni = 0; ni < 4; ++ni) {
      int col = col0 + ni * 16 + (lane & 15);
#pragma unroll
      for (int v = 0; v < 8; ++v) {
        int row = row0 + mi * 16 + rbase + v;
        out[(size_t)row * D + col] = acc[mi][ni][v] + resid[(size_t)row * D + col];
      }
    }
}

// ---------------------------------------------------------------------------
extern "C" void kernel_launch(void* const* d_in, const int* in_sizes, int n_in,
                              void* d_out, int out_size, void* d_ws, size_t ws_size,
                              hipStream_t stream) {
  const float* h1 = (const float*)d_in[0];
  const float* h2 = (const float*)d_in[1];
  const float* Wq = (const float*)d_in[2];
  const float* bq = (const float*)d_in[3];
  const float* Wk = (const float*)d_in[4];
  const float* bk = (const float*)d_in[5];

  float* outp   = (float*)d_out;
  float* fused1 = outp;                                // [N, D]
  float* fused2 = outp + (size_t)NROW * DDIM;          // [M, D]
  float* attn   = outp + 2ull * NROW * DDIM;           // [N, M]

  // workspace layout (~67.2 MB)
  char* ws = (char*)d_ws;
  float*  q_f32 = (float*)ws;            ws += (size_t)NROW * DDIM * 4;
  float*  k_f32 = (float*)ws;            ws += (size_t)MROW * DDIM * 4;
  __bf16* q_bf  = (__bf16*)ws;           ws += (size_t)NROW * DDIM * 2;
  __bf16* k_bf  = (__bf16*)ws;           ws += (size_t)MROW * DDIM * 2;
  __bf16* qT_bf = (__bf16*)ws;           ws += (size_t)NROW * DDIM * 2;
  __bf16* kT_bf = (__bf16*)ws;           ws += (size_t)MROW * DDIM * 2;
  float*  rmax  = (float*)ws;            ws += (size_t)NROW * 4;
  float*  rrinv = (float*)ws;            ws += (size_t)NROW * 4;
  float*  cmax  = (float*)ws;            ws += (size_t)MROW * 4;
  float*  crinv = (float*)ws;            ws += (size_t)MROW * 4;

  // 1) projections (q, k) + bf16 / bf16^T copies
  proj_kernel<<<NROW / 64, 256, 0, stream>>>(h1, Wq, bq, q_f32, q_bf, qT_bf, NROW);
  proj_kernel<<<MROW / 64, 256, 0, stream>>>(h2, Wk, bk, k_f32, k_bf, kT_bf, MROW);
  // 2) attn = q @ k^T / sqrt(D)  -> straight into d_out
  attn_kernel<<<dim3(NROW / 64, MROW / 512), 256, 0, stream>>>(q_bf, k_bf, attn);
  // 3) softmax stats
  rowstats_kernel<<<NROW, 256, 0, stream>>>(attn, rmax, rrinv);
  colstats_kernel<<<MROW / 256, 256, 0, stream>>>(attn, cmax, crinv);
  // 4) fused1 = softmax_row(attn) @ k + q ; fused2 = softmax_col(attn)^T @ q + k
  fused_kernel<<<NROW / 64, 256, 0, stream>>>(attn, rmax, rrinv, kT_bf, q_f32, fused1, 0);
  fused_kernel<<<MROW / 64, 256, 0, stream>>>(attn, cmax, crinv, qT_bf, k_f32, fused2, 1);
}
